// MAHGCNNET_82248623718658
// MI455X (gfx1250) — matrix operand order repaired and checked by reading, not compiled, not run on hardware
//
#include <hip/hip_runtime.h>
#include <hip/hip_bf16.h>

// MAHGCN forward on MI455X (gfx1250): all contractions via v_wmma_f32_16x16x32_f16.
// Wave32; 128-thread blocks = 4 waves; block tile 64(M) x 64(N), K-step 32.
// Each wave computes a 32x32 output tile: 2 A-frags x 2 B-frags = 4 WMMA /
// k-step with 4 independent accumulator chains (covers the 5-slot WMMA RAW
// hazard window). LDS tiles are fragment-major: every fragment load is a
// contiguous 32-byte LDS read; every staging store is a ds_store_b128.

typedef _Float16 f16;
typedef __attribute__((ext_vector_type(16))) _Float16 v16h;
typedef __attribute__((ext_vector_type(8)))  _Float16 v8h;
typedef __attribute__((ext_vector_type(4)))  _Float16 v4h;
typedef __attribute__((ext_vector_type(8)))  float    v8f;

constexpr int BM  = 64;  // block tile M (4 m-tiles; 2 per wave)
constexpr int BNN = 64;  // block tile N (4 n-tiles; 2 per wave)
constexpr int BK  = 32;  // K step per WMMA

// -------- vectorized 8-element row loads (fast path; lda % 4 == 0 holds) ----
__device__ __forceinline__ void load8(const float* __restrict__ p, float v[8]) {
    const float4 a = ((const float4*)p)[0];
    const float4 b = ((const float4*)p)[1];
    v[0] = a.x; v[1] = a.y; v[2] = a.z; v[3] = a.w;
    v[4] = b.x; v[5] = b.y; v[6] = b.z; v[7] = b.w;
}
__device__ __forceinline__ void load8(const f16* __restrict__ p, float v[8]) {
    const v4h a = ((const v4h*)p)[0];
    const v4h b = ((const v4h*)p)[1];
    v[0] = (float)a.x; v[1] = (float)a.y; v[2] = (float)a.z; v[3] = (float)a.w;
    v[4] = (float)b.x; v[5] = (float)b.y; v[6] = (float)b.z; v[7] = (float)b.w;
}

// fragment-major LDS offsets (halves)
__device__ __forceinline__ int aOff(int r, int kq) {   // r: row in tile, kq: k-octet
    return (((r >> 4) * 32) + (r & 15) + ((kq & 1) << 4)) * 16 + ((kq >> 1) << 3);
}
__device__ __forceinline__ int bOff(int c, int kq) {   // c: col in tile, kq: k-octet
    return (((c >> 4) * 32) + (c & 15) + ((kq & 1) << 4)) * 16 + ((kq >> 1) << 3);
}

// ---------------------------------------------------------------------------
// Templated WMMA GEMM:  C[b] = op( A[b] @ B[b] + bias ),  A:[M,K] B:[K,N] row-major
// ---------------------------------------------------------------------------
template <typename TA, typename TB, typename TC, bool BIAS, bool RELU>
__global__ __launch_bounds__(128) void gemm_wmma(
    const TA* __restrict__ A, long long strideA, int lda,
    const TB* __restrict__ B, long long strideB, int ldb,
    TC* __restrict__ C, long long strideC, int ldc,
    const float* __restrict__ bias, int M, int N, int K)
{
    __shared__ __attribute__((aligned(32))) _Float16 As[4 * 32 * 16]; // 64x32, 4KB
    __shared__ __attribute__((aligned(32))) _Float16 Bs[4 * 32 * 16]; // 32x64, 4KB

    const int bz = blockIdx.z;
    const int m0 = blockIdx.y * BM;
    const int n0 = blockIdx.x * BNN;
    A += (long long)bz * strideA;
    B += (long long)bz * strideB;
    C += (long long)bz * strideC;

    const int t    = threadIdx.x;          // 0..127
    const int lane = t & 31;
    const int w    = t >> 5;               // 0..3
    const int wm   = w & 1;                // m super-tile (32 rows)
    const int wn   = w >> 1;               // n super-tile (32 cols)

    // staging maps (16 elems/thread for each operand, 2 octets of 8)
    const int ar   = t >> 1;               // A row in tile, 0..63
    const int akqb = (t & 1) * 2;          // A k-octet base (0 or 2)
    const int bc   = t & 63;               // B col in tile, 0..63
    const int bkqb = (t >> 6) * 2;         // B k-octet base (0 or 2)

    v8f acc00 = {}, acc01 = {}, acc10 = {}, acc11 = {};

    for (int k0 = 0; k0 < K; k0 += BK) {
        // ================= stage A (64x32) =================
        if (m0 + BM <= M && k0 + BK <= K) {          // uniform fast path
            #pragma unroll
            for (int q = 0; q < 2; ++q) {
                const int kq = akqb + q;
                const TA* src = A + (long long)(m0 + ar) * lda + k0 + kq * 8;
                float v[8];
                load8(src, v);
                if (q == 0 && k0 + BK < K) __builtin_prefetch(src + BK, 0, 1);
                v8h pk;
                #pragma unroll
                for (int e = 0; e < 8; ++e) pk[e] = (_Float16)v[e];
                *(v8h*)&As[aOff(ar, kq)] = pk;
            }
        } else {                                      // branchless clamped path
            #pragma unroll
            for (int q = 0; q < 2; ++q) {
                const int kq = akqb + q;
                v8h pk;
                #pragma unroll
                for (int e = 0; e < 8; ++e) {
                    int gm = m0 + ar, gk = k0 + kq * 8 + e;
                    int cm = gm < M ? gm : M - 1;
                    int ck = gk < K ? gk : K - 1;
                    float v = (float)A[(long long)cm * lda + ck];
                    pk[e] = (_Float16)((gm < M && gk < K) ? v : 0.f);
                }
                *(v8h*)&As[aOff(ar, kq)] = pk;
            }
        }
        // ================= stage B (32x64) =================
        if (n0 + BNN <= N && k0 + BK <= K) {          // uniform fast path
            #pragma unroll
            for (int q = 0; q < 2; ++q) {
                const int kq = bkqb + q;
                const TB* src = B + (long long)(k0 + kq * 8) * ldb + n0 + bc;
                float v[8];
                #pragma unroll
                for (int e = 0; e < 8; ++e) v[e] = (float)src[(long long)e * ldb];
                if (q == 0 && k0 + BK < K) __builtin_prefetch(src + (long long)BK * ldb, 0, 1);
                v8h pk;
                #pragma unroll
                for (int e = 0; e < 8; ++e) pk[e] = (_Float16)v[e];
                *(v8h*)&Bs[bOff(bc, kq)] = pk;
            }
        } else {                                      // branchless clamped path
            #pragma unroll
            for (int q = 0; q < 2; ++q) {
                const int kq = bkqb + q;
                v8h pk;
                #pragma unroll
                for (int e = 0; e < 8; ++e) {
                    int gk = k0 + kq * 8 + e, gn = n0 + bc;
                    int ck = gk < K ? gk : K - 1;
                    int cn = gn < N ? gn : N - 1;
                    float v = (float)B[(long long)ck * ldb + cn];
                    pk[e] = (_Float16)((gk < K && gn < N) ? v : 0.f);
                }
                *(v8h*)&Bs[bOff(bc, kq)] = pk;
            }
        }
        __syncthreads();

        const v16h a0 = *(const v16h*)&As[((2 * wm + 0) * 32 + lane) * 16];
        const v16h a1 = *(const v16h*)&As[((2 * wm + 1) * 32 + lane) * 16];
        const v16h b0 = *(const v16h*)&Bs[((2 * wn + 0) * 32 + lane) * 16];
        const v16h b1 = *(const v16h*)&Bs[((2 * wn + 1) * 32 + lane) * 16];
        acc00 = __builtin_amdgcn_wmma_f32_16x16x32_f16(false, a0, false, b0, (short)0, acc00, false, false);
        acc01 = __builtin_amdgcn_wmma_f32_16x16x32_f16(false, a0, false, b1, (short)0, acc01, false, false);
        acc10 = __builtin_amdgcn_wmma_f32_16x16x32_f16(false, a1, false, b0, (short)0, acc10, false, false);
        acc11 = __builtin_amdgcn_wmma_f32_16x16x32_f16(false, a1, false, b1, (short)0, acc11, false, false);
        __syncthreads();
    }

    // ---- epilogue: C/D layout — lane gives col, vgpr gives row ----
    #pragma unroll
    for (int i = 0; i < 2; ++i) {
        const int rbase = m0 + (2 * wm + i) * 16 + ((lane >> 4) << 3);
        #pragma unroll
        for (int j = 0; j < 2; ++j) {
            const int col = n0 + (2 * wn + j) * 16 + (lane & 15);
            const v8f& acc = i ? (j ? acc11 : acc10) : (j ? acc01 : acc00);
            if (col < N) {
                float bv = BIAS ? bias[col] : 0.f;
                #pragma unroll
                for (int v = 0; v < 8; ++v) {
                    int row = rbase + v;
                    if (row < M) {
                        float x = acc[v] + bv;
                        if (RELU) x = x > 0.f ? x : 0.f;
                        C[(long long)row * ldc + col] = (TC)x;
                    }
                }
            }
        }
    }
}

// ---------------------------------------------------------------------------
// Small helper kernels
// ---------------------------------------------------------------------------
__global__ void diag_kernel(const f16* __restrict__ h, int N,
                            float* __restrict__ feats, int off)
{
    int i = blockIdx.x * blockDim.x + threadIdx.x;
    if (i >= 256 * N) return;
    int b = i / N, n = i % N;
    feats[(size_t)b * 1500 + off + n] = (float)h[((size_t)b * N + n) * N + n];
}

__global__ void bn_stats_kernel(const float* __restrict__ x, int F,
                                float* __restrict__ mean, float* __restrict__ var)
{
    int f = blockIdx.x * blockDim.x + threadIdx.x;
    if (f >= F) return;
    float s = 0.f, s2 = 0.f;
    for (int r = 0; r < 256; ++r) {
        float v = x[(size_t)r * F + f];
        s += v; s2 += v * v;
    }
    float m = s * (1.f / 256.f);
    mean[f] = m;
    var[f]  = s2 * (1.f / 256.f) - m * m;   // biased var, matches jnp.var
}

__global__ void bn_apply_kernel(const float* __restrict__ x,
                                const float* __restrict__ mean,
                                const float* __restrict__ var,
                                const float* __restrict__ gamma,
                                const float* __restrict__ beta,
                                f16* __restrict__ y, int F)
{
    int i = blockIdx.x * blockDim.x + threadIdx.x;
    if (i >= 256 * F) return;
    int f = i % F;
    float v = gamma[f] * (x[i] - mean[f]) * rsqrtf(var[f] + 1e-5f) + beta[f];
    y[i] = (f16)(v > 0.f ? v : 0.f);
}

__global__ void softmax2_kernel(const float* __restrict__ z, float* __restrict__ out)
{
    int i = blockIdx.x * blockDim.x + threadIdx.x;
    if (i >= 256) return;
    float a = z[2 * i], b = z[2 * i + 1];
    float m = fmaxf(a, b);
    float ea = __expf(a - m), eb = __expf(b - m);
    float inv = 1.f / (ea + eb);
    out[2 * i] = ea * inv;
    out[2 * i + 1] = eb * inv;
}

// ---------------------------------------------------------------------------
// Host-side launch helpers
// ---------------------------------------------------------------------------
template <typename TA, typename TB, typename TC, bool BIAS, bool RELU>
static inline void launch_gemm(const void* A, long long sA, int lda,
                               const void* B, long long sB, int ldb,
                               void* C, long long sC, int ldc,
                               const float* bias, int M, int N, int K, int batch,
                               hipStream_t stream)
{
    dim3 grid((N + BNN - 1) / BNN, (M + BM - 1) / BM, batch);
    gemm_wmma<TA, TB, TC, BIAS, RELU><<<grid, dim3(128), 0, stream>>>(
        (const TA*)A, sA, lda, (const TB*)B, sB, ldb, (TC*)C, sC, ldc, bias, M, N, K);
}

static inline void launch_diag(const f16* h, int N, float* feats, int off, hipStream_t s)
{
    diag_kernel<<<(256 * N + 255) / 256, 256, 0, s>>>(h, N, feats, off);
}

extern "C" void kernel_launch(void* const* d_in, const int* in_sizes, int n_in,
                              void* d_out, int out_size, void* d_ws, size_t ws_size,
                              hipStream_t stream)
{
    (void)in_sizes; (void)n_in; (void)out_size; (void)ws_size;

    // ---- inputs (setup_inputs dict order) ----
    const float* g[5];
    for (int i = 0; i < 5; ++i) g[i] = (const float*)d_in[i];
    const float* W[5];  const float* bvec[5];
    for (int i = 0; i < 5; ++i) { W[i] = (const float*)d_in[5 + 2 * i]; bvec[i] = (const float*)d_in[6 + 2 * i]; }
    const float* Mx[4];
    for (int i = 0; i < 4; ++i) Mx[i] = (const float*)d_in[15 + i];
    const float* flW[4]; const float* flb[4]; const float* gam[4]; const float* bet[4];
    for (int i = 0; i < 4; ++i) {
        flW[i] = (const float*)d_in[19 + 4 * i];
        flb[i] = (const float*)d_in[20 + 4 * i];
        gam[i] = (const float*)d_in[21 + 4 * i];
        bet[i] = (const float*)d_in[22 + 4 * i];
    }

    // ---- workspace carve-out ----
    char* ws = (char*)d_ws;
    size_t off = 0;
    auto take = [&](size_t bytes) -> void* {
        void* p = ws + off;
        off = (off + bytes + 255) & ~(size_t)255;
        return p;
    };
    f16*   X     = (f16*)take(256ULL * 500 * 500 * sizeof(f16));   // 128 MB ping
    f16*   Y     = (f16*)take(256ULL * 400 * 500 * sizeof(f16));   // 102 MB pong
    float* feats = (float*)take(256ULL * 1500 * sizeof(float));
    f16*   bnbuf = (f16*)take(256ULL * 1500 * sizeof(f16));
    float* zA    = (float*)take(256ULL * 512 * sizeof(float));
    float* zB    = (float*)take(256ULL * 512 * sizeof(float));
    float* meanb = (float*)take(1500 * sizeof(float));
    float* varb  = (float*)take(1500 * sizeof(float));

    const int Bn = 256;
    const int NS[5] = {500, 400, 300, 200, 100};

    // ===== Layer 1: h = identity, so h1 = relu(g1 @ W1 + b1) -> X [B,500,500]
    launch_gemm<float, float, f16, true, true>(
        g[0], 500LL * 500, 500, W[0], 0, 500, X, 500LL * 500, 500,
        bvec[0], 500, 500, 500, Bn, stream);
    launch_diag(X, 500, feats, 0, stream);

    // ===== Layers 2..5: pool -> t = p @ W -> h = relu(g @ t + b) -> diag
    const f16* hcur = X;
    int foff = 500;
    for (int i = 1; i < 5; ++i) {
        int Np = NS[i - 1];   // previous node count (rows of h_prev)
        int Nc = NS[i];       // current node count
        int Fp = NS[i - 1];   // feature width of h_prev (square)
        f16* pbuf = (hcur == X) ? Y : X;
        // p = M_i @ h_prev : [Nc,Np]@[B,Np,Fp] -> [B,Nc,Fp]
        launch_gemm<float, f16, f16, false, false>(
            Mx[i - 1], 0, Np, hcur, (long long)Np * Fp, Fp,
            pbuf, (long long)Nc * Fp, Fp, nullptr, Nc, Fp, Np, Bn, stream);
        // t = p @ W_i : [B,Nc,Fp]@[Fp,Nc] -> [B,Nc,Nc]
        f16* tbuf = (pbuf == X) ? Y : X;
        launch_gemm<f16, float, f16, false, false>(
            pbuf, (long long)Nc * Fp, Fp, W[i], 0, Nc,
            tbuf, (long long)Nc * Nc, Nc, nullptr, Nc, Nc, Fp, Bn, stream);
        // h = relu(g_i @ t + b_i) : [B,Nc,Nc]
        f16* hbuf = (tbuf == X) ? Y : X;
        launch_gemm<float, f16, f16, true, true>(
            g[i], (long long)Nc * Nc, Nc, tbuf, (long long)Nc * Nc, Nc,
            hbuf, (long long)Nc * Nc, Nc, bvec[i], Nc, Nc, Nc, Bn, stream);
        launch_diag(hbuf, Nc, feats, foff, stream);
        foff += Nc;
        hcur = hbuf;
    }

    // ===== MLP: 4x [BN(batch-stats)+ReLU -> Linear], then softmax
    const int dims[5] = {1500, 512, 256, 128, 2};
    float* act = feats;
    for (int i = 0; i < 4; ++i) {
        int F = dims[i], O = dims[i + 1];
        bn_stats_kernel<<<(F + 255) / 256, 256, 0, stream>>>(act, F, meanb, varb);
        bn_apply_kernel<<<(256 * F + 255) / 256, 256, 0, stream>>>(
            act, meanb, varb, gam[i], bet[i], bnbuf, F);
        float* next = (i & 1) ? zB : zA;
        if (act == next) next = (next == zA) ? zB : zA;  // safety
        launch_gemm<f16, float, float, true, false>(
            bnbuf, 0, F, flW[i], 0, O, next, 0, O, flb[i], 256, O, F, 1, stream);
        act = next;
    }
    softmax2_kernel<<<1, 256, 0, stream>>>(act, (float*)d_out);
}
